// P2PDecoderLayer_71116068487655
// MI455X (gfx1250) — matrix operand order repaired and checked by reading, not compiled, not run on hardware
//
#include <hip/hip_runtime.h>

#define DD 256
#define HH 8
#define FF 1024
#define PP 16384
#define SS 2048
#define NN 1024
#define KEE 2048
#define KNN 2048
#define BB 2
#define BD (BB * DD)

typedef __attribute__((ext_vector_type(8)))  __bf16 v8bf;
typedef __attribute__((ext_vector_type(16))) __bf16 v16bf;
typedef __attribute__((ext_vector_type(8)))  float  v8f;

// ---------------- device helpers ----------------

// A-fragment (16-bit): lane<16 holds K {0..15}, lane>=16 holds K {16..31}
// caller passes p already offset by (half*16); we fetch [p..p+15]
__device__ __forceinline__ v16bf fragA(const __bf16* p) {
  v8bf lo = *(const v8bf*)p;
  v8bf hi = *(const v8bf*)(p + 8);
  return __builtin_shufflevector(lo, hi, 0,1,2,3,4,5,6,7,8,9,10,11,12,13,14,15);
}
// B-fragment (16-bit): lane<16 holds K {0..15}, lane>=16 holds K {16..31}
// caller passes p already offset by (half*16); contiguous 16 elements
__device__ __forceinline__ v16bf fragB(const __bf16* p) {
  v8bf lo = *(const v8bf*)p;
  v8bf hi = *(const v8bf*)(p + 8);
  return __builtin_shufflevector(lo, hi, 0,1,2,3,4,5,6,7,8,9,10,11,12,13,14,15);
}

__device__ __forceinline__ float redsum32(float v) {
#pragma unroll
  for (int m = 16; m >= 1; m >>= 1) v += __shfl_xor(v, m, 32);
  return v;
}
__device__ __forceinline__ float redmax16(float v) {
#pragma unroll
  for (int m = 8; m >= 1; m >>= 1) v = fmaxf(v, __shfl_xor(v, m, 32));
  return v;
}
__device__ __forceinline__ float redsum16(float v) {
#pragma unroll
  for (int m = 8; m >= 1; m >>= 1) v += __shfl_xor(v, m, 32);
  return v;
}

// gfx1250 async global->LDS copy (tracked with ASYNCcnt), per CDNA5 ISA.
// lds_off: byte address in LDS (low 32 bits of a generic pointer to __shared__),
// gaddr: 64-bit global address; 16 bytes per lane.
__device__ __forceinline__ void async_copy_b128(unsigned lds_off, const void* gaddr) {
  asm volatile("global_load_async_to_lds_b128 %0, %1, off"
               :: "v"(lds_off), "v"(gaddr) : "memory");
}
__device__ __forceinline__ void wait_async_le1() {
  asm volatile("s_wait_asynccnt 0x1" ::: "memory");
}
__device__ __forceinline__ void wait_async_le0() {
  asm volatile("s_wait_asynccnt 0x0" ::: "memory");
}

// ---------------- kernels ----------------

// C[M,N] = A[M,K] @ W[N,K]^T + bias (+relu). A,W bf16 row-major; C f32.
// Block tile 64(M) x 128(N), 8 waves, each wave 32x32 (4 accumulators).
// A panel (64x32 per K-step) is staged through LDS with double-buffered
// global_load_async_to_lds_b128; W streams from global (hits L2: weights are
// <1 MB vs 192 MB L2). Requires M%64==0, N%128==0, K%32==0 (true for all
// shapes in this layer).
__global__ __launch_bounds__(256)
void gemm_bf16_kernel(const __bf16* __restrict__ A, const __bf16* __restrict__ W,
                      const float* __restrict__ bias, float* __restrict__ C,
                      int M, int N, int K, int relu) {
  __shared__ __align__(16) __bf16 As[2][64 * 32];
  int tid = threadIdx.x;
  int wave = tid >> 5;
  int lane = tid & 31;
  int half = lane >> 4;
  int r = lane & 15;
  int mw = wave >> 2;   // 0..1
  int nw = wave & 3;    // 0..3
  int mblk = blockIdx.x * 64;
  int n0 = blockIdx.y * 128 + nw * 32;

  // cooperative A staging: thread -> (row 0..63, 8-elem K chunk 0..3)
  int arow = tid >> 2;
  int akq = tid & 3;
  const __bf16* aga = A + (size_t)(mblk + arow) * K + akq * 8;
  unsigned ldsA0 = (unsigned)(size_t)(&As[0][0]) + (unsigned)((arow * 32 + akq * 8) * 2);
  unsigned ldsA1 = (unsigned)(size_t)(&As[1][0]) + (unsigned)((arow * 32 + akq * 8) * 2);

  int nk = K >> 5;
  async_copy_b128(ldsA0, aga);
  if (nk > 1) async_copy_b128(ldsA1, aga + 32);

  const __bf16* w0 = W + (size_t)(n0 + r) * K + half * 16;
  const __bf16* w1 = W + (size_t)(n0 + 16 + r) * K + half * 16;

  v8f acc00 = {0,0,0,0,0,0,0,0};
  v8f acc01 = {0,0,0,0,0,0,0,0};
  v8f acc10 = {0,0,0,0,0,0,0,0};
  v8f acc11 = {0,0,0,0,0,0,0,0};

  const __bf16* af0 = &As[0][(mw * 32 + r) * 32 + half * 16];
  const __bf16* af1 = &As[0][(mw * 32 + 16 + r) * 32 + half * 16];

  for (int i = 0; i < nk; ++i) {
    int k0 = i << 5;
    if (i + 1 < nk) wait_async_le1(); else wait_async_le0();
    __syncthreads();
    int boff = (i & 1) * (64 * 32);
    v16bf a0 = fragA(af0 + boff);
    v16bf a1 = fragA(af1 + boff);
    v16bf b0 = fragB(w0 + k0);
    v16bf b1 = fragB(w1 + k0);
    acc00 = __builtin_amdgcn_wmma_f32_16x16x32_bf16(false, a0, false, b0, (short)0, acc00, false, false);
    acc01 = __builtin_amdgcn_wmma_f32_16x16x32_bf16(false, a0, false, b1, (short)0, acc01, false, false);
    acc10 = __builtin_amdgcn_wmma_f32_16x16x32_bf16(false, a1, false, b0, (short)0, acc10, false, false);
    acc11 = __builtin_amdgcn_wmma_f32_16x16x32_bf16(false, a1, false, b1, (short)0, acc11, false, false);
    __syncthreads();
    if (i + 2 < nk) {
      async_copy_b128((i & 1) ? ldsA1 : ldsA0, aga + (size_t)k0 + 64);
      __builtin_prefetch(w0 + k0 + 64, 0, 1);
      __builtin_prefetch(w1 + k0 + 64, 0, 1);
    }
  }

  float bb0 = bias ? bias[n0 + r] : 0.0f;
  float bb1 = bias ? bias[n0 + 16 + r] : 0.0f;
#pragma unroll
  for (int mi = 0; mi < 2; ++mi) {
    int mbase = mblk + mw * 32 + mi * 16;
    v8f aN = mi ? acc10 : acc00;
    v8f aP = mi ? acc11 : acc01;
#pragma unroll
    for (int g = 0; g < 8; ++g) {
      int row = mbase + g + half * 8;
      float v0 = aN[g] + bb0;
      float v1 = aP[g] + bb1;
      if (relu) { v0 = fmaxf(v0, 0.0f); v1 = fmaxf(v1, 0.0f); }
      C[(size_t)row * N + n0 + r] = v0;
      C[(size_t)row * N + n0 + 16 + r] = v1;
    }
  }
}

// Flash-attention: one wave per 16-query tile, stream keys in chunks of 32.
// Q/K/V are bf16 with row index (l*BB + b) and given element strides (3D for
// fused QKV, 2D for fused KV, D for standalone Q); head h occupies cols h*32..h*32+31.
// Output f32 [Lq, B, D].
__global__ __launch_bounds__(128)
void attn_kernel(const __bf16* __restrict__ qb, int qStride,
                 const __bf16* __restrict__ kb, int kStride,
                 const __bf16* __restrict__ vb, int vStride,
                 float* __restrict__ out, int Lk) {
  __shared__ __align__(16) __bf16 pbuf[4][16 * 32];
  int wave = threadIdx.x >> 5;
  int lane = threadIdx.x & 31;
  int half = lane >> 4;
  int r = lane & 15;
  int q0 = (blockIdx.x * 4 + wave) << 4;
  int b = blockIdx.y / HH;
  int h = blockIdx.y % HH;
  const float scale = 0.17677669529663687f;  // 1/sqrt(32)

  v16bf qf = fragA(qb + ((size_t)(q0 + r) * BB + b) * qStride + h * 32 + half * 16);

  v8f acc0 = {0,0,0,0,0,0,0,0};
  v8f acc1 = {0,0,0,0,0,0,0,0};
  float rowmax[8], rowsum[8];
#pragma unroll
  for (int g = 0; g < 8; ++g) { rowmax[g] = -3.0e38f; rowsum[g] = 0.0f; }

  __bf16* pb = &pbuf[wave][0];

  for (int k0 = 0; k0 < Lk; k0 += 32) {
    // scores: A=Q(16x32 dh), B=K^T(32x16 keys): lane = key, K-dim = dh (contiguous)
    v16bf kf0 = fragB(kb + ((size_t)(k0 + r) * BB + b) * kStride + h * 32 + half * 16);
    v16bf kf1 = fragB(kb + ((size_t)(k0 + 16 + r) * BB + b) * kStride + h * 32 + half * 16);
    v8f s0 = {0,0,0,0,0,0,0,0};
    v8f s1 = {0,0,0,0,0,0,0,0};
    s0 = __builtin_amdgcn_wmma_f32_16x16x32_bf16(false, qf, false, kf0, (short)0, s0, false, false);
    s1 = __builtin_amdgcn_wmma_f32_16x16x32_bf16(false, qf, false, kf1, (short)0, s1, false, false);

    float p0[8], p1[8];
#pragma unroll
    for (int g = 0; g < 8; ++g) {
      float a0 = s0[g] * scale, a1 = s1[g] * scale;
      float mt = redmax16(fmaxf(a0, a1));            // row max within 16-lane group
      float mnew = fmaxf(rowmax[g], mt);
      float corr = __expf(rowmax[g] - mnew);
      float e0 = __expf(a0 - mnew);
      float e1 = __expf(a1 - mnew);
      float ps = redsum16(e0 + e1);
      rowsum[g] = rowsum[g] * corr + ps;
      rowmax[g] = mnew;
      acc0[g] *= corr;
      acc1[g] *= corr;
      p0[g] = e0; p1[g] = e1;
    }
    // transpose P from C-layout registers to A-layout via LDS (row-major 16x32)
#pragma unroll
    for (int g = 0; g < 8; ++g) {
      pb[(g + half * 8) * 32 + r] = (__bf16)p0[g];
      pb[(g + half * 8) * 32 + 16 + r] = (__bf16)p1[g];
    }
    __syncthreads();
    v16bf pf = fragA(pb + r * 32 + half * 16);
    // V fragments: lane = dh col, K-dim = key (strided)
    v16bf vf0, vf1;
#pragma unroll
    for (int i = 0; i < 16; ++i) {
      size_t kro = ((size_t)(k0 + half * 16 + i) * BB + b) * vStride + h * 32;
      vf0[i] = vb[kro + r];
      vf1[i] = vb[kro + 16 + r];
    }
    acc0 = __builtin_amdgcn_wmma_f32_16x16x32_bf16(false, pf, false, vf0, (short)0, acc0, false, false);
    acc1 = __builtin_amdgcn_wmma_f32_16x16x32_bf16(false, pf, false, vf1, (short)0, acc1, false, false);
    __syncthreads();
  }
#pragma unroll
  for (int g = 0; g < 8; ++g) {
    float inv = 1.0f / rowsum[g];
    size_t o = ((size_t)(q0 + g + half * 8) * BB + b) * DD + h * 32;
    out[o + r] = acc0[g] * inv;
    out[o + 16 + r] = acc1[g] * inv;
  }
}

// out = LN(x (+r1) (+r2)) over rows of length 256; one wave per row.
__global__ __launch_bounds__(256)
void ln_kernel(const float* __restrict__ x, const float* __restrict__ r1,
               const float* __restrict__ r2, const float* __restrict__ g,
               const float* __restrict__ bta, float* __restrict__ out, int rows) {
  int wave = threadIdx.x >> 5;
  int lane = threadIdx.x & 31;
  int row = blockIdx.x * 8 + wave;
  if (row >= rows) return;
  size_t base = (size_t)row * DD;
  float v[8];
  float s = 0.0f;
#pragma unroll
  for (int i = 0; i < 8; ++i) {
    int idx = lane + i * 32;
    float t = x[base + idx];
    if (r1) t += r1[base + idx];
    if (r2) t += r2[base + idx];
    v[i] = t;
    s += t;
  }
  s = redsum32(s);
  float mean = s * (1.0f / DD);
  float vs = 0.0f;
#pragma unroll
  for (int i = 0; i < 8; ++i) { float d = v[i] - mean; vs += d * d; }
  vs = redsum32(vs);
  float rstd = rsqrtf(vs * (1.0f / DD) + 1e-5f);
#pragma unroll
  for (int i = 0; i < 8; ++i) {
    int idx = lane + i * 32;
    out[base + idx] = (v[i] - mean) * rstd * g[idx] + bta[idx];
  }
}

__global__ __launch_bounds__(256)
void cvt_kernel(const float* __restrict__ in, __bf16* __restrict__ out, size_t n) {
  size_t i = (size_t)blockIdx.x * blockDim.x + threadIdx.x;
  size_t stride = (size_t)gridDim.x * blockDim.x;
  for (; i < n; i += stride) out[i] = (__bf16)in[i];
}

__global__ __launch_bounds__(256)
void gather_rows_kernel(const float* __restrict__ table, const int* __restrict__ idx,
                        float* __restrict__ out) {
  int i = blockIdx.x;
  size_t row = (size_t)idx[i];
  for (int e = threadIdx.x; e < BD; e += 256)
    out[(size_t)i * BD + e] = table[row * BD + e];
}

__global__ __launch_bounds__(256)
void scatter_rows_kernel(float* __restrict__ table, const int* __restrict__ idx,
                         const float* __restrict__ src) {
  int i = blockIdx.x;
  size_t row = (size_t)idx[i];
  for (int e = threadIdx.x; e < BD; e += 256)
    table[row * BD + e] = src[(size_t)i * BD + e];
}

__global__ __launch_bounds__(256)
void scatter_zero_kernel(float* __restrict__ table, const int* __restrict__ idx) {
  int i = blockIdx.x;
  size_t row = (size_t)idx[i];
  for (int e = threadIdx.x; e < BD; e += 256)
    table[row * BD + e] = 0.0f;
}

// ---------------- host helpers ----------------

struct Bump {
  char* base;
  size_t cur;
  void* get(size_t bytes) {
    cur = (cur + 255) & ~(size_t)255;
    void* p = base + cur;
    cur += bytes;
    return p;
  }
};

static void launch_gemm(hipStream_t st, const __bf16* A, const __bf16* W, const float* bias,
                        float* C, int M, int N, int K, int relu) {
  gemm_bf16_kernel<<<dim3(M / 64, N / 128), dim3(256), 0, st>>>(A, W, bias, C, M, N, K, relu);
}
static void launch_cvt(hipStream_t st, const float* in, __bf16* out, size_t n) {
  int blocks = (int)((n + 1023) / 1024);
  if (blocks > 2048) blocks = 2048;
  if (blocks < 1) blocks = 1;
  cvt_kernel<<<dim3(blocks), dim3(256), 0, st>>>(in, out, n);
}
static void launch_ln(hipStream_t st, const float* x, const float* r1, const float* r2,
                      const float* g, const float* b, float* out, int rows) {
  ln_kernel<<<dim3((rows + 7) / 8), dim3(256), 0, st>>>(x, r1, r2, g, b, out, rows);
}
static void launch_attn(hipStream_t st, const __bf16* q, int qs, const __bf16* k, int ks,
                        const __bf16* v, int vs, float* out, int Lq, int Lk) {
  attn_kernel<<<dim3(Lq / 64, BB * HH), dim3(128), 0, st>>>(q, qs, k, ks, v, vs, out, Lk);
}

// self attention: fused QKV projection [rows,3D], flash attention, out-proj
static void mha_self(hipStream_t st, char* scratch, const __bf16* xb, int L,
                     const float* bi, const __bf16* wib, const __bf16* wob,
                     const float* bo, float* out) {
  Bump s{scratch, 0};
  int rows = L * BB;
  float*  qkv  = (float*)s.get((size_t)rows * 3 * DD * 4);
  __bf16* qkvb = (__bf16*)s.get((size_t)rows * 3 * DD * 2);
  float*  ao   = (float*)s.get((size_t)rows * DD * 4);
  __bf16* aob  = (__bf16*)s.get((size_t)rows * DD * 2);
  launch_gemm(st, xb, wib, bi, qkv, rows, 3 * DD, DD, 0);
  launch_cvt(st, qkv, qkvb, (size_t)rows * 3 * DD);
  launch_attn(st, qkvb, 3 * DD, qkvb + DD, 3 * DD, qkvb + 2 * DD, 3 * DD, ao, L, L);
  launch_cvt(st, ao, aob, (size_t)rows * DD);
  launch_gemm(st, aob, wob, bo, out, rows, DD, DD, 0);
}

// cross attention: Q proj from q input (wi rows 0..D), KV proj from kv input (wi rows D..3D)
static void mha_cross(hipStream_t st, char* scratch, const __bf16* qxb, int Lq,
                      const __bf16* kvxb, int Lk, const float* bi, const __bf16* wib,
                      const __bf16* wob, const float* bo, float* out) {
  Bump s{scratch, 0};
  int qr = Lq * BB, kr = Lk * BB;
  float*  qf   = (float*)s.get((size_t)qr * DD * 4);
  __bf16* qfb  = (__bf16*)s.get((size_t)qr * DD * 2);
  float*  kvf  = (float*)s.get((size_t)kr * 2 * DD * 4);
  __bf16* kvfb = (__bf16*)s.get((size_t)kr * 2 * DD * 2);
  float*  ao   = (float*)s.get((size_t)qr * DD * 4);
  __bf16* aob  = (__bf16*)s.get((size_t)qr * DD * 2);
  launch_gemm(st, qxb, wib, bi, qf, qr, DD, DD, 0);
  launch_gemm(st, kvxb, wib + (size_t)DD * DD, bi + DD, kvf, kr, 2 * DD, DD, 0);
  launch_cvt(st, qf, qfb, (size_t)qr * DD);
  launch_cvt(st, kvf, kvfb, (size_t)kr * 2 * DD);
  launch_attn(st, qfb, DD, kvfb, 2 * DD, kvfb + DD, 2 * DD, ao, Lq, Lk);
  launch_cvt(st, ao, aob, (size_t)qr * DD);
  launch_gemm(st, aob, wob, bo, out, qr, DD, DD, 0);
}

// out = LN(x + relu(x@W1^T+b1)@W2^T+b2)
static void ffn_block(hipStream_t st, char* scratch, const float* x, int rows,
                      const __bf16* w1b, const float* b1, const __bf16* w2b, const float* b2,
                      const float* gg, const float* bbeta, float* out) {
  Bump s{scratch, 0};
  __bf16* xb = (__bf16*)s.get((size_t)rows * DD * 2);
  float*  h  = (float*)s.get((size_t)rows * FF * 4);
  __bf16* hb = (__bf16*)s.get((size_t)rows * FF * 2);
  float*  f  = (float*)s.get((size_t)rows * DD * 4);
  launch_cvt(st, x, xb, (size_t)rows * DD);
  launch_gemm(st, xb, w1b, b1, h, rows, FF, DD, 1);
  launch_cvt(st, h, hb, (size_t)rows * FF);
  launch_gemm(st, hb, w2b, b2, f, rows, DD, FF, 0);
  launch_ln(st, x, f, nullptr, gg, bbeta, out, rows);
}

// ---------------- entry point ----------------

extern "C" void kernel_launch(void* const* d_in, const int* in_sizes, int n_in,
                              void* d_out, int out_size, void* d_ws, size_t ws_size,
                              hipStream_t stream) {
  (void)in_sizes; (void)n_in; (void)out_size;
  const float* sp_in = (const float*)d_in[0];
  const float* su_in = (const float*)d_in[1];
  const float* ep    = (const float*)d_in[2];
  const float *wi[6], *bi[6], *wo[6], *bo[6];
  for (int m = 0; m < 6; ++m) {  // sa, san, e2e, e2n, n2e, n2n
    wi[m] = (const float*)d_in[3 + 4 * m];
    bi[m] = (const float*)d_in[4 + 4 * m];
    wo[m] = (const float*)d_in[5 + 4 * m];
    bo[m] = (const float*)d_in[6 + 4 * m];
  }
  const float* n1g  = (const float*)d_in[27]; const float* n1b  = (const float*)d_in[28];
  const float* n2g  = (const float*)d_in[29]; const float* n2b  = (const float*)d_in[30];
  const float* n3g  = (const float*)d_in[31]; const float* n3b  = (const float*)d_in[32];
  const float* n1ug = (const float*)d_in[33]; const float* n1ub = (const float*)d_in[34];
  const float* n2ug = (const float*)d_in[35]; const float* n2ub = (const float*)d_in[36];
  const float* n3ug = (const float*)d_in[37]; const float* n3ub = (const float*)d_in[38];
  const float* l1w  = (const float*)d_in[39]; const float* l1bs = (const float*)d_in[40];
  const float* l2w  = (const float*)d_in[41]; const float* l2bs = (const float*)d_in[42];
  const float* l1uw = (const float*)d_in[43]; const float* l1ubs = (const float*)d_in[44];
  const float* l2uw = (const float*)d_in[45]; const float* l2ubs = (const float*)d_in[46];
  const int* ind_pair = (const int*)d_in[47];
  const int* ind_e2e  = (const int*)d_in[48];
  const int* ind_n2e  = (const int*)d_in[49];

  const size_t SBD = (size_t)SS * BB * DD;
  const size_t NBD = (size_t)NN * BB * DD;
  const size_t PBD = (size_t)PP * BB * DD;
  const size_t KEBD = (size_t)KEE * BB * DD;
  const size_t KNBD = (size_t)KNN * BB * DD;

  float* out_unary = (float*)d_out;
  float* out_pair  = out_unary + NBD;
  float* out_e2e   = out_pair + SBD;
  float* out_e2n   = out_e2e + SBD;

  Bump a{(char*)d_ws, 0};
  __bf16 *wib[6], *wob[6];
  for (int m = 0; m < 6; ++m) {
    wib[m] = (__bf16*)a.get((size_t)3 * DD * DD * 2);
    wob[m] = (__bf16*)a.get((size_t)DD * DD * 2);
  }
  __bf16* l1b_  = (__bf16*)a.get((size_t)FF * DD * 2);
  __bf16* l2b_  = (__bf16*)a.get((size_t)DD * FF * 2);
  __bf16* l1ub_ = (__bf16*)a.get((size_t)FF * DD * 2);
  __bf16* l2ub_ = (__bf16*)a.get((size_t)DD * FF * 2);
  __bf16* spb = (__bf16*)a.get(SBD * 2);
  __bf16* sub = (__bf16*)a.get(NBD * 2);
  float* att_sp = (float*)a.get(SBD * 4);
  float* att_su = (float*)a.get(NBD * 4);
  float* sp_ln = (float*)a.get(SBD * 4);
  __bf16* sp_lnb = (__bf16*)a.get(SBD * 2);
  float* su_ln = (float*)a.get(NBD * 4);
  __bf16* su_lnb = (__bf16*)a.get(NBD * 2);
  float* pair0 = (float*)a.get(PBD * 4);
  float* pair1 = (float*)a.get(PBD * 4);
  float* pn2e = (float*)a.get(KNBD * 4);
  __bf16* pn2eb = (__bf16*)a.get(KNBD * 2);
  float* pe2e = (float*)a.get(KEBD * 4);
  __bf16* pe2eb = (__bf16*)a.get(KEBD * 2);
  float* att_e2e = (float*)a.get(SBD * 4);
  float* att_e2n = (float*)a.get(SBD * 4);
  float* att_n2e = (float*)a.get(NBD * 4);
  float* att_n2n = (float*)a.get(NBD * 4);
  float* x1 = (float*)a.get(SBD * 4);
  float* x2 = (float*)a.get(SBD * 4);
  float* x3 = (float*)a.get(SBD * 4);
  float* uu = (float*)a.get(NBD * 4);
  char* scratch = (char*)a.get((size_t)40 << 20);
  if (a.cur > ws_size) return;  // workspace too small; bail without OOB writes

  // weight conversion f32 -> bf16
  for (int m = 0; m < 6; ++m) {
    launch_cvt(stream, wi[m], wib[m], (size_t)3 * DD * DD);
    launch_cvt(stream, wo[m], wob[m], (size_t)DD * DD);
  }
  launch_cvt(stream, l1w, l1b_, (size_t)FF * DD);
  launch_cvt(stream, l2w, l2b_, (size_t)DD * FF);
  launch_cvt(stream, l1uw, l1ub_, (size_t)FF * DD);
  launch_cvt(stream, l2uw, l2ub_, (size_t)DD * FF);

  // ---- self-attention + norm1 (pair & unary) ----
  launch_cvt(stream, sp_in, spb, SBD);
  mha_self(stream, scratch, spb, SS, bi[0], wib[0], wob[0], bo[0], att_sp);
  launch_ln(stream, sp_in, att_sp, nullptr, n1g, n1b, sp_ln, SS * BB);
  launch_cvt(stream, sp_ln, sp_lnb, SBD);

  launch_cvt(stream, su_in, sub, NBD);
  mha_self(stream, scratch, sub, NN, bi[1], wib[1], wob[1], bo[1], att_su);
  launch_ln(stream, su_in, att_su, nullptr, n1ug, n1ub, su_ln, NN * BB);
  launch_cvt(stream, su_ln, su_lnb, NBD);

  // ---- pair tables: pair0 = entire_pair with ind_pair rows zeroed ----
  (void)hipMemcpyAsync(pair0, ep, PBD * 4, hipMemcpyDeviceToDevice, stream);
  scatter_zero_kernel<<<dim3(SS), dim3(256), 0, stream>>>(pair0, ind_pair);
  gather_rows_kernel<<<dim3(KNN), dim3(256), 0, stream>>>(pair0, ind_n2e, pn2e);
  launch_cvt(stream, pn2e, pn2eb, KNBD);
  // pair1 = pair0 with sp_ln scattered at ind_pair; gather at ind_e2e
  (void)hipMemcpyAsync(pair1, pair0, PBD * 4, hipMemcpyDeviceToDevice, stream);
  scatter_rows_kernel<<<dim3(SS), dim3(256), 0, stream>>>(pair1, ind_pair, sp_ln);
  gather_rows_kernel<<<dim3(KEE), dim3(256), 0, stream>>>(pair1, ind_e2e, pe2e);
  launch_cvt(stream, pe2e, pe2eb, KEBD);

  // ---- cross attentions on sp (each computed once, reused across branches) ----
  mha_cross(stream, scratch, sp_lnb, SS, pe2eb, KEE, bi[2], wib[2], wob[2], bo[2], att_e2e);
  mha_cross(stream, scratch, sp_lnb, SS, su_lnb, NN, bi[3], wib[3], wob[3], bo[3], att_e2n);

  // ---- three update() branches ----
  launch_ln(stream, sp_ln, att_e2e, nullptr, n2g, n2b, x1, SS * BB);
  ffn_block(stream, scratch, x1, SS * BB, l1b_, l1bs, l2b_, l2bs, n3g, n3b, out_e2e);
  launch_ln(stream, sp_ln, att_e2n, nullptr, n2g, n2b, x2, SS * BB);
  ffn_block(stream, scratch, x2, SS * BB, l1b_, l1bs, l2b_, l2bs, n3g, n3b, out_e2n);
  launch_ln(stream, sp_ln, att_e2e, att_e2n, n2g, n2b, x3, SS * BB);
  ffn_block(stream, scratch, x3, SS * BB, l1b_, l1bs, l2b_, l2bs, n3g, n3b, out_pair);

  // ---- unary update ----
  mha_cross(stream, scratch, su_lnb, NN, pn2eb, KNN, bi[4], wib[4], wob[4], bo[4], att_n2e);
  mha_self(stream, scratch, su_lnb, NN, bi[5], wib[5], wob[5], bo[5], att_n2n);
  launch_ln(stream, su_ln, att_n2e, att_n2n, n2ug, n2ub, uu, NN * BB);
  ffn_block(stream, scratch, uu, NN * BB, l1ub_, l1ubs, l2ub_, l2ubs, n3ug, n3ub, out_unary);
}
